// GPTDecoder_48378511622359
// MI455X (gfx1250) — compile-verified, matrix operand order checked by reading, not run on hardware
//
#include <hip/hip_runtime.h>

// ---------------- model constants (match reference) ----------------
#define DMODEL 1024
#define NHEAD  16
#define HDIM   64
#define SEQ    1024
#define NB     2
#define NLAYER 8
#define VOCAB  32000
#define MTOT   (NB * SEQ)   // 2048 rows

typedef __attribute__((ext_vector_type(16))) __bf16 v16bf;
typedef __attribute__((ext_vector_type(8)))  __bf16 v8bf;
typedef __attribute__((ext_vector_type(8)))  float  v8f;
typedef __attribute__((ext_vector_type(4)))  int    v4i;

// ---------------- CDNA5 async global->LDS staging (ASYNCcnt-tracked) ----------------
#if defined(__has_builtin)
#if __has_builtin(__builtin_amdgcn_global_load_async_to_lds_b128) && \
    __has_builtin(__builtin_amdgcn_s_wait_asynccnt)
#define USE_ASYNC_LDS 1
#endif
#endif
#ifndef USE_ASYNC_LDS
#define USE_ASYNC_LDS 0
#endif

// per-lane 16-byte copy: global -> LDS
__device__ __forceinline__ void stage16(const __bf16* g, __bf16* l) {
#if USE_ASYNC_LDS
    v4i* gv = (v4i*)g;    // strip const / retype in generic AS
    v4i* lv = (v4i*)l;
    __builtin_amdgcn_global_load_async_to_lds_b128(
        (__attribute__((address_space(1))) v4i*)gv,
        (__attribute__((address_space(3))) v4i*)lv, 0, 0);
#else
    *(uint4*)l = *(const uint4*)g;
#endif
}
__device__ __forceinline__ void stage_join() {
#if USE_ASYNC_LDS
    __builtin_amdgcn_s_wait_asynccnt(0);
#endif
}

// ---------------- WMMA helpers ----------------
// A/B fragment layout for V_WMMA_F32_16X16X32_BF16 (ISA 7.12.2):
//  lanes 0-15  : row (A) / col (B) = lane, K chunks {0..7} then {16..23}
//  lanes 16-31 : same row/col,      K chunks {8..15} then {24..31}
__device__ __forceinline__ v16bf ldfrag(const __bf16* p, int row, int ld, int k0, int half) {
    const __bf16* q = p + (size_t)row * ld + k0 + half * 8;
    v8bf lo = *(const v8bf*)q;
    v8bf hi = *(const v8bf*)(q + 16);
    return __builtin_shufflevector(lo, hi, 0,1,2,3,4,5,6,7,8,9,10,11,12,13,14,15);
}

__device__ __forceinline__ v8f wmma_bf16(v16bf a, v16bf b, v8f c) {
    // (neg_a, A, neg_b, B, c_mod, C, reuse_a, reuse_b)
    return __builtin_amdgcn_wmma_f32_16x16x32_bf16(false, a, false, b, (short)0, c, false, false);
}

// ---------------- embedding ----------------
__global__ __launch_bounds__(256) void embed_kernel(const int* __restrict__ ids,
                                                    const float* __restrict__ temb,
                                                    const float* __restrict__ pemb,
                                                    float* __restrict__ x) {
    int row = blockIdx.x;            // 0..2047  (b*SEQ + t)
    int t   = row & (SEQ - 1);
    int id  = ids[row];
    int d   = threadIdx.x * 4;
    float4 a = *(const float4*)(temb + (size_t)id * DMODEL + d);
    float4 p = *(const float4*)(pemb + (size_t)t  * DMODEL + d);
    float4 r; r.x = a.x + p.x; r.y = a.y + p.y; r.z = a.z + p.z; r.w = a.w + p.w;
    *(float4*)(x + (size_t)row * DMODEL + d) = r;
}

// ---------------- layernorm (fp32 in -> bf16 out) ----------------
__global__ __launch_bounds__(256) void ln_kernel(const float* __restrict__ x,
                                                 const float* __restrict__ g,
                                                 const float* __restrict__ b,
                                                 __bf16* __restrict__ out) {
    int row = blockIdx.x;
    int tid = threadIdx.x;
    float4 v = *(const float4*)(x + (size_t)row * DMODEL + tid * 4);
    float s = v.x + v.y + v.z + v.w;
    float q = v.x * v.x + v.y * v.y + v.z * v.z + v.w * v.w;
    __shared__ float rs[256], rq[256];
    rs[tid] = s; rq[tid] = q;
    __syncthreads();
    for (int st = 128; st > 0; st >>= 1) {
        if (tid < st) { rs[tid] += rs[tid + st]; rq[tid] += rq[tid + st]; }
        __syncthreads();
    }
    float mean = rs[0] * (1.0f / DMODEL);
    float var  = rq[0] * (1.0f / DMODEL) - mean * mean;   // biased, matches reference
    float rstd = rsqrtf(var + 1e-5f);
    int d = tid * 4;
    float4 gg = *(const float4*)(g + d);
    float4 bb = *(const float4*)(b + d);
    __bf16* o = out + (size_t)row * DMODEL + d;
    o[0] = (__bf16)((v.x - mean) * rstd * gg.x + bb.x);
    o[1] = (__bf16)((v.y - mean) * rstd * gg.y + bb.y);
    o[2] = (__bf16)((v.z - mean) * rstd * gg.z + bb.z);
    o[3] = (__bf16)((v.w - mean) * rstd * gg.w + bb.w);
}

// ---------------- GEMM: out[M,N] = A[M,K](bf16) * W[N,K]^T(fp32) + epilogue ----------------
constexpr int EPI_BF16 = 0;   // +bias -> bf16 out
constexpr int EPI_GELU = 1;   // +bias, exact gelu -> bf16 out
constexpr int EPI_RES  = 2;   // +bias, +residual -> fp32 out
constexpr int EPI_NONE = 3;   // plain -> fp32 out

template <int EPI>
__global__ __launch_bounds__(256) void gemm_bf16_kernel(
    const __bf16* __restrict__ A, const float* __restrict__ W,
    const float* __restrict__ bias, const float* __restrict__ residual,
    float* __restrict__ outF, __bf16* __restrict__ outH,
    int M, int N, int K) {
    (void)M;
    __shared__ __align__(16) __bf16 As[128 * 32];   // [m][k]
    __shared__ __align__(16) __bf16 Bs[128 * 32];   // [n][k]
    int tid  = threadIdx.x;
    int lane = tid & 31, wave = tid >> 5;
    int l16 = lane & 15, hh = lane >> 4;
    int wm = wave >> 1, wn = wave & 1;              // 4 x 2 wave grid
    int bm = blockIdx.y * 128, bn = blockIdx.x * 128;
    v8f acc[2][4] = {};

    int r = tid >> 1, s16 = (tid & 1) * 16;
    const __bf16* gA = A + (size_t)(bm + r) * K + s16;
    const float*  gW = W + (size_t)(bn + r) * K + s16;

    for (int k0 = 0; k0 < K; k0 += 32) {
        // stage A tile: bf16 straight copy, async global->LDS when available
        stage16(gA + k0,     As + r * 32 + s16);
        stage16(gA + k0 + 8, As + r * 32 + s16 + 8);
        // prefetch next weight slice (global_prefetch_b8)
        if (k0 + 32 < K) __builtin_prefetch(gW + k0 + 32, 0, 0);
        {   // stage W tile: fp32 -> bf16 on the fly (VALU convert, LDS store)
            const float* g = gW + k0;
            __bf16* dl = Bs + r * 32 + s16;
            #pragma unroll
            for (int j = 0; j < 16; j += 4) {
                float4 f = *(const float4*)(g + j);
                dl[j + 0] = (__bf16)f.x; dl[j + 1] = (__bf16)f.y;
                dl[j + 2] = (__bf16)f.z; dl[j + 3] = (__bf16)f.w;
            }
        }
        stage_join();
        __syncthreads();
        v16bf af[2], bfg[4];
        #pragma unroll
        for (int mi = 0; mi < 2; ++mi) af[mi]  = ldfrag(As, wm * 32 + mi * 16 + l16, 32, 0, hh);
        #pragma unroll
        for (int ni = 0; ni < 4; ++ni) bfg[ni] = ldfrag(Bs, wn * 64 + ni * 16 + l16, 32, 0, hh);
        #pragma unroll
        for (int mi = 0; mi < 2; ++mi)
            #pragma unroll
            for (int ni = 0; ni < 4; ++ni)
                acc[mi][ni] = wmma_bf16(af[mi], bfg[ni], acc[mi][ni]);
        __syncthreads();
    }

    // epilogue; C/D layout: VGPR i -> row i (+8 for lanes 16..31), col = lane%16
    #pragma unroll
    for (int mi = 0; mi < 2; ++mi) {
        #pragma unroll
        for (int ni = 0; ni < 4; ++ni) {
            int col = bn + wn * 64 + ni * 16 + l16;
            float bv = (EPI == EPI_NONE) ? 0.f : bias[col];
            #pragma unroll
            for (int i = 0; i < 8; ++i) {
                int row = bm + wm * 32 + mi * 16 + i + hh * 8;
                float v = acc[mi][ni][i] + bv;
                if (EPI == EPI_GELU)
                    v = 0.5f * v * (1.0f + erff(v * 0.70710678118654752f));
                size_t idx = (size_t)row * N + col;
                if (EPI == EPI_RES)       { v += residual[idx]; outF[idx] = v; }
                else if (EPI == EPI_NONE) { outF[idx] = v; }
                else                      { outH[idx] = (__bf16)v; }
            }
        }
    }
}

// ---------------- flash attention, causal, hd=64 ----------------
// grid: (SEQ/64, NHEAD, NB); 128 threads = 4 waves; wave w owns q rows [w*16, w*16+16)
__global__ __launch_bounds__(128) void attn_kernel(const __bf16* __restrict__ qkv,
                                                   __bf16* __restrict__ o) {
    int qt = blockIdx.x, head = blockIdx.y, b = blockIdx.z;
    int tid = threadIdx.x;
    int lane = tid & 31, wave = tid >> 5;
    int l16 = lane & 15, hh = lane >> 4;
    __shared__ __align__(16) __bf16 Ks[64 * 64];     // [key][d]
    __shared__ __align__(16) __bf16 Vt[64 * 64];     // [d][key]  (transposed)
    __shared__ __align__(16) __bf16 Ps[4][16 * 64];  // per-wave P tile

    const int ldq = 3 * DMODEL;
    int q0 = qt * 64;
    size_t rowbase = (size_t)b * SEQ;

    // Q fragments (held in registers for the whole kernel)
    const __bf16* qbase = qkv + (rowbase + q0 + wave * 16) * ldq + head * HDIM;
    v16bf qf0 = ldfrag(qbase, l16, ldq, 0,  hh);
    v16bf qf1 = ldfrag(qbase, l16, ldq, 32, hh);

    v8f oacc[4] = {};
    float m_i[8], l_i[8];
    #pragma unroll
    for (int i = 0; i < 8; ++i) { m_i[i] = -__builtin_inff(); l_i[i] = 0.f; }

    int kr = tid >> 1, ks16 = (tid & 1) * 32;   // K-tile staging coords

    for (int kt = 0; kt <= qt; ++kt) {
        int k0g = kt * 64;
        {   // stage K tile [64 keys][64 d]: 64B per thread, async when available
            const __bf16* g = qkv + (rowbase + k0g + kr) * ldq + DMODEL + head * HDIM + ks16;
            __bf16* dl = Ks + kr * 64 + ks16;
            stage16(g,      dl);
            stage16(g + 8,  dl + 8);
            stage16(g + 16, dl + 16);
            stage16(g + 24, dl + 24);
        }
        // stage V transposed: Vt[d][key]
        for (int idx = tid; idx < 64 * 64; idx += 128) {
            int key = idx >> 6, d = idx & 63;
            Vt[d * 64 + key] = qkv[(rowbase + k0g + key) * ldq + 2 * DMODEL + head * HDIM + d];
        }
        stage_join();
        __syncthreads();

        // S = Q K^T (16 x 64 per wave): preload all 8 fragments, then 8 WMMAs
        v16bf kb[8];
        #pragma unroll
        for (int ks = 0; ks < 2; ++ks)
            #pragma unroll
            for (int ni = 0; ni < 4; ++ni)
                kb[ks * 4 + ni] = ldfrag(Ks, ni * 16 + l16, 64, ks * 32, hh);
        v8f sacc[4] = {};
        #pragma unroll
        for (int ks = 0; ks < 2; ++ks) {
            v16bf a = ks ? qf1 : qf0;
            #pragma unroll
            for (int ni = 0; ni < 4; ++ni)
                sacc[ni] = wmma_bf16(a, kb[ks * 4 + ni], sacc[ni]);
        }

        // online softmax (per lane: 8 rows; columns spread over 16-lane group)
        float pv[4][8];
        #pragma unroll
        for (int i = 0; i < 8; ++i) {
            int row = q0 + wave * 16 + i + hh * 8;
            float mx = -__builtin_inff();
            #pragma unroll
            for (int ni = 0; ni < 4; ++ni) {
                int col = k0g + ni * 16 + l16;
                float v = (col <= row) ? sacc[ni][i] * 0.125f : -__builtin_inff();
                pv[ni][i] = v;
                mx = fmaxf(mx, v);
            }
            mx = fmaxf(mx, __shfl_xor(mx, 1, 16));
            mx = fmaxf(mx, __shfl_xor(mx, 2, 16));
            mx = fmaxf(mx, __shfl_xor(mx, 4, 16));
            mx = fmaxf(mx, __shfl_xor(mx, 8, 16));
            float mnew  = fmaxf(m_i[i], mx);
            float alpha = __expf(m_i[i] - mnew);
            m_i[i] = mnew;
            float rs = 0.f;
            #pragma unroll
            for (int ni = 0; ni < 4; ++ni) {
                float p = __expf(pv[ni][i] - mnew);
                pv[ni][i] = p;
                rs += p;
            }
            rs += __shfl_xor(rs, 1, 16);
            rs += __shfl_xor(rs, 2, 16);
            rs += __shfl_xor(rs, 4, 16);
            rs += __shfl_xor(rs, 8, 16);
            l_i[i] = l_i[i] * alpha + rs;
            #pragma unroll
            for (int ni = 0; ni < 4; ++ni) oacc[ni][i] *= alpha;
        }

        // spill P (C layout) to LDS so it can be re-read in A-fragment layout
        #pragma unroll
        for (int ni = 0; ni < 4; ++ni)
            #pragma unroll
            for (int i = 0; i < 8; ++i)
                Ps[wave][(i + hh * 8) * 64 + ni * 16 + l16] = (__bf16)pv[ni][i];
        __syncthreads();

        // O += P V: preload P and V fragments, then 8 back-to-back WMMAs
        v16bf pa0 = ldfrag(Ps[wave], l16, 64, 0,  hh);
        v16bf pa1 = ldfrag(Ps[wave], l16, 64, 32, hh);
        v16bf vb[8];
        #pragma unroll
        for (int ks = 0; ks < 2; ++ks)
            #pragma unroll
            for (int ni = 0; ni < 4; ++ni)
                vb[ks * 4 + ni] = ldfrag(Vt, ni * 16 + l16, 64, ks * 32, hh);
        #pragma unroll
        for (int ks = 0; ks < 2; ++ks) {
            v16bf a = ks ? pa1 : pa0;
            #pragma unroll
            for (int ni = 0; ni < 4; ++ni)
                oacc[ni] = wmma_bf16(a, vb[ks * 4 + ni], oacc[ni]);
        }
        __syncthreads();
    }

    // normalize and write merged-head output [B*T, D] bf16
    #pragma unroll
    for (int ni = 0; ni < 4; ++ni)
        #pragma unroll
        for (int i = 0; i < 8; ++i) {
            int row = q0 + wave * 16 + i + hh * 8;
            int col = head * HDIM + ni * 16 + l16;
            o[(rowbase + row) * DMODEL + col] = (__bf16)(oacc[ni][i] / l_i[i]);
        }
}

// ---------------- host orchestration ----------------
extern "C" void kernel_launch(void* const* d_in, const int* in_sizes, int n_in,
                              void* d_out, int out_size, void* d_ws, size_t ws_size,
                              hipStream_t stream) {
    (void)in_sizes; (void)n_in; (void)out_size; (void)ws_size;
    const int*   ids   = (const int*)  d_in[0];
    const float* temb  = (const float*)d_in[1];
    const float* pemb  = (const float*)d_in[2];
    const float* qkv_w = (const float*)d_in[3];
    const float* qkv_b = (const float*)d_in[4];
    const float* out_w = (const float*)d_in[5];
    const float* out_b = (const float*)d_in[6];
    const float* ln1_w = (const float*)d_in[7];
    const float* ln1_b = (const float*)d_in[8];
    const float* ln2_w = (const float*)d_in[9];
    const float* ln2_b = (const float*)d_in[10];
    const float* w1    = (const float*)d_in[11];
    const float* b1    = (const float*)d_in[12];
    const float* w2    = (const float*)d_in[13];
    const float* b2    = (const float*)d_in[14];
    const float* lnf_w = (const float*)d_in[15];
    const float* lnf_b = (const float*)d_in[16];
    const float* lmw   = (const float*)d_in[17];

    char* ws = (char*)d_ws;
    float*  x   = (float*)(ws);                     //  8 MiB  fp32 residual stream
    __bf16* h   = (__bf16*)(ws + (8u  << 20));      //  4 MiB  ln output
    __bf16* qkv = (__bf16*)(ws + (12u << 20));      // 12 MiB  qkv bf16
    __bf16* ob  = (__bf16*)(ws + (24u << 20));      //  4 MiB  attention out
    __bf16* ff  = (__bf16*)(ws + (28u << 20));      // 16 MiB  ffn hidden

    embed_kernel<<<MTOT, 256, 0, stream>>>(ids, temb, pemb, x);

    for (int l = 0; l < NLAYER; ++l) {
        ln_kernel<<<MTOT, 256, 0, stream>>>(x, ln1_w + l * DMODEL, ln1_b + l * DMODEL, h);
        gemm_bf16_kernel<EPI_BF16><<<dim3(3 * DMODEL / 128, MTOT / 128), 256, 0, stream>>>(
            h, qkv_w + (size_t)l * 3 * DMODEL * DMODEL, qkv_b + (size_t)l * 3 * DMODEL,
            nullptr, nullptr, qkv, MTOT, 3 * DMODEL, DMODEL);
        attn_kernel<<<dim3(SEQ / 64, NHEAD, NB), 128, 0, stream>>>(qkv, ob);
        gemm_bf16_kernel<EPI_RES><<<dim3(DMODEL / 128, MTOT / 128), 256, 0, stream>>>(
            ob, out_w + (size_t)l * DMODEL * DMODEL, out_b + l * DMODEL,
            x, x, nullptr, MTOT, DMODEL, DMODEL);
        ln_kernel<<<MTOT, 256, 0, stream>>>(x, ln2_w + l * DMODEL, ln2_b + l * DMODEL, h);
        gemm_bf16_kernel<EPI_GELU><<<dim3(4 * DMODEL / 128, MTOT / 128), 256, 0, stream>>>(
            h, w1 + (size_t)l * 4 * DMODEL * DMODEL, b1 + (size_t)l * 4 * DMODEL,
            nullptr, nullptr, ff, MTOT, 4 * DMODEL, DMODEL);
        gemm_bf16_kernel<EPI_RES><<<dim3(DMODEL / 128, MTOT / 128), 256, 0, stream>>>(
            ff, w2 + (size_t)l * DMODEL * 4 * DMODEL, b2 + l * DMODEL,
            x, x, nullptr, MTOT, DMODEL, 4 * DMODEL);
    }

    ln_kernel<<<MTOT, 256, 0, stream>>>(x, lnf_w, lnf_b, h);
    gemm_bf16_kernel<EPI_NONE><<<dim3(VOCAB / 128, MTOT / 128), 256, 0, stream>>>(
        h, lmw, nullptr, nullptr, (float*)d_out, nullptr, MTOT, VOCAB, DMODEL);
}
